// _CustomLinear_54803782697499
// MI455X (gfx1250) — compile-verified, matrix operand order checked by reading, not run on hardware
//
#include <hip/hip_runtime.h>
#include <hip/hip_bf16.h>

typedef __attribute__((ext_vector_type(16))) _Float16 v16h;
typedef __attribute__((ext_vector_type(8)))  float    v8f;

#define K_DIM 4096
#define N_DIM 14336

// Load the 16 f32 values a lane owns for one 16x16x32 WMMA chunk:
// two contiguous 8-float runs at p[0..7] and p[16..23]  (32B-aligned each).
__device__ __forceinline__ void ld16(const float* __restrict__ p, float* dst) {
    const float4 a = *reinterpret_cast<const float4*>(p);
    const float4 b = *reinterpret_cast<const float4*>(p + 4);
    const float4 c = *reinterpret_cast<const float4*>(p + 16);
    const float4 d = *reinterpret_cast<const float4*>(p + 20);
    dst[0]=a.x;  dst[1]=a.y;  dst[2]=a.z;  dst[3]=a.w;
    dst[4]=b.x;  dst[5]=b.y;  dst[6]=b.z;  dst[7]=b.w;
    dst[8]=c.x;  dst[9]=c.y;  dst[10]=c.z; dst[11]=c.w;
    dst[12]=d.x; dst[13]=d.y; dst[14]=d.z; dst[15]=d.w;
}

// Block = 256 threads = 8 waves. Block handles 16 output features (one WMMA
// N-tile); waves split K 8x512 = 2 quant-groups of 256 each. Wave 0 sums the 8 partials via LDS.
__global__ __launch_bounds__(256) void qc4w_linear_kernel(
    const float* __restrict__ x,      // [16, 4096]
    const float* __restrict__ w,      // [14336, 4096]
    float* __restrict__ out)          // [16, 14336]
{
    const int lane  = threadIdx.x & 31;
    const int wave  = threadIdx.x >> 5;
    const int lmod  = lane & 15;
    const int khalf = (lane & 16) ? 8 : 0;     // K-half owned by this lane
    const int n0    = blockIdx.x * 16;

    const float* __restrict__ wrow = w + (size_t)(n0 + lmod) * K_DIM; // B col = weight row
    const float* __restrict__ xrow = x + (size_t)lmod * K_DIM;        // A row = seq pos

    v8f acc = {};   // f32 output fragment for this wave's K-slice

#pragma unroll
    for (int g = 0; g < 2; ++g) {
        const int kbase = wave * 512 + g * 256;   // one quant group (256 along K)

        // ---- pass 1: stage this lane's half-group (128 f32) in VGPRs ----
        float wbuf[128];
#pragma unroll
        for (int c = 0; c < 8; ++c) {
            ld16(wrow + kbase + c * 32 + khalf, &wbuf[c * 16]);
        }

        // ---- group max|w|, combined with partner lane (other K-half) ----
        float m = 0.0f;
#pragma unroll
        for (int i = 0; i < 128; ++i) m = fmaxf(m, fabsf(wbuf[i]));
        m = fmaxf(m, __shfl_xor(m, 16, 32));                 // wave32 partner
        const float scale = fmaxf(m * (1.0f / 7.0f), 1e-9f); // maxabs/QMAX, eps
        const float inv   = 1.0f / scale;

        // ---- pass 2: quantize to exact-int4-in-f16, 8 chained WMMAs ----
        v8f gacc = {};
#pragma unroll
        for (int c = 0; c < 8; ++c) {
            float xv[16];
            ld16(xrow + kbase + c * 32 + khalf, xv);
            v16h a, b;
#pragma unroll
            for (int j = 0; j < 16; ++j) {
                // round-half-even (v_rndne) to match jnp.round, clamp [-8,7]
                float q = fminf(fmaxf(rintf(wbuf[c * 16 + j] * inv), -8.0f), 7.0f);
                b[j] = (_Float16)q;       // exact in f16
                a[j] = (_Float16)xv[j];   // only lossy conversion
            }
            gacc = __builtin_amdgcn_wmma_f32_16x16x32_f16(
                false, a, false, b, (short)0, gacc, false, false);
        }

        // D layout: lane owns column N = lane%16  ==  weight row n0+lmod,
        // so the per-(row,group) scale is a per-lane scalar.
#pragma unroll
        for (int e = 0; e < 8; ++e) acc[e] += scale * gacc[e];
    }

    // ---- deterministic cross-wave reduction (fixed order, no atomics) ----
    __shared__ float red[8][32][8];     // 8 KB
#pragma unroll
    for (int e = 0; e < 8; ++e) red[wave][lane][e] = acc[e];
    __syncthreads();

    if (wave == 0) {
#pragma unroll
        for (int e = 0; e < 8; ++e) {
            float s = 0.0f;
#pragma unroll
            for (int p = 0; p < 8; ++p) s += red[p][lane][e];
            const int M = e + ((lane & 16) ? 8 : 0);     // C/D VGPR->row map
            out[(size_t)M * N_DIM + n0 + lmod] = s;
        }
    }
}

extern "C" void kernel_launch(void* const* d_in, const int* in_sizes, int n_in,
                              void* d_out, int out_size, void* d_ws, size_t ws_size,
                              hipStream_t stream) {
    const float* x = (const float*)d_in[0];   // [1,16,4096] f32
    const float* w = (const float*)d_in[1];   // [14336,4096] f32
    float* out     = (float*)d_out;           // [1,16,14336] f32

    dim3 grid(N_DIM / 16);   // 896 tiles of 16 output features
    dim3 block(256);         // 8 waves
    qc4w_linear_kernel<<<grid, block, 0, stream>>>(x, w, out);
}
